// LossAwareMemoryBank_20486994002594
// MI455X (gfx1250) — matrix operand order repaired and checked by reading, not compiled, not run on hardware
//
#include <hip/hip_runtime.h>
#include <hip/hip_bf16.h>

// ---------------------------------------------------------------------------
// LossAwareMemoryBank on MI455X (gfx1250, wave32)
//
//   1) lamb_normalize_bf16: L2-normalize rows of query (4096x1024) and
//      memory (65536x1024), store bf16 in d_ws.  The bf16 bank (128 MB)
//      fits in the 192 MB L2, so the similarity GEMM streams from L2.
//   2) lamb_sim_topk: 16 query rows per workgroup; stream the bank in
//      256-row chunks.  Each of the 8 waves computes two 16x16 WMMA tiles
//      sharing one A fragment (v_wmma_f32_16x16x32_bf16 with reuse_a).
//      Streaming top-10 uses a threshold filter: 256 threads scan against
//      the row's current 10th-best (LDS), survivors go through ds-atomic
//      compaction, and 16 owner threads do sorted insertion.  Epilogue:
//      adaptive-k masked softmax + weighted gather of raw f32 bank rows.
// ---------------------------------------------------------------------------

typedef __attribute__((ext_vector_type(16))) __bf16 v16bf;
typedef __attribute__((ext_vector_type(8)))  float  v8f;

#define FEAT_DIM    1024
#define BANK_SIZE   65536
#define BATCH       4096
#define NUM_CLASSES 14
#define QB          16     // query rows per workgroup (one WMMA M tile)
#define NB          256    // bank rows per chunk (8 waves x 2 tiles x 16)
#define MAXK        10
#define CAP         64     // candidate list capacity per row per chunk

// ---------------------------------------------------------------------------
// Row L2-normalize + convert to bf16.  One 256-thread block per row.
// ---------------------------------------------------------------------------
__global__ __launch_bounds__(256) void lamb_normalize_bf16(
    const float* __restrict__ in, __bf16* __restrict__ out)
{
    __shared__ float red[256];
    const int row = blockIdx.x;
    const int tid = threadIdx.x;

    const float4 x = ((const float4*)(in + (size_t)row * FEAT_DIM))[tid];
    red[tid] = x.x * x.x + x.y * x.y + x.z * x.z + x.w * x.w;
    __syncthreads();
    for (int off = 128; off > 0; off >>= 1) {
        if (tid < off) red[tid] += red[tid + off];
        __syncthreads();
    }
    const float inv = 1.0f / fmaxf(sqrtf(red[0]), 1e-12f);

    union { __bf16 h[4]; uint2 u; } pk;
    pk.h[0] = (__bf16)(x.x * inv);
    pk.h[1] = (__bf16)(x.y * inv);
    pk.h[2] = (__bf16)(x.z * inv);
    pk.h[3] = (__bf16)(x.w * inv);
    ((uint2*)(out + (size_t)row * FEAT_DIM))[tid] = pk.u;
}

// ---------------------------------------------------------------------------
// Main kernel.  Grid: BATCH/QB = 256 blocks, 256 threads (8 wave32s).
// ---------------------------------------------------------------------------
__global__ __launch_bounds__(256) void lamb_sim_topk(
    const __bf16* __restrict__ qn,      // normalized query, bf16 [BATCH][D]
    const __bf16* __restrict__ mn,      // normalized bank,  bf16 [BANK][D]
    const float*  __restrict__ memory,  // raw bank, f32 [BANK][D]
    const float*  __restrict__ preds,   // [BATCH][NUM_CLASSES]
    float*        __restrict__ out)     // [BATCH][D]
{
    __shared__ __align__(16) __bf16 As[QB * FEAT_DIM];   // 32 KB A tile
    __shared__ float sim[QB][NB + 4];                    // 16.6 KB sim tile
    __shared__ float thr[QB];                            // row 10th-best
    __shared__ int   scnt[QB];                           // candidate counts
    __shared__ float candv[QB][CAP];
    __shared__ int   candi[QB][CAP];
    __shared__ float wbuf[QB][MAXK];
    __shared__ int   ibuf[QB][MAXK];

    const int tid   = threadIdx.x;
    const int wave  = tid >> 5;
    const int lane  = tid & 31;
    const int l15   = lane & 15;
    const int lhalf = lane >> 4;                  // 0: lanes 0-15, 1: 16-31
    const int qbase = blockIdx.x * QB;

    // ---- Stage the 16x1024 bf16 query tile into LDS (2048 uint4) ----------
    {
        const uint4* src = (const uint4*)(qn + (size_t)qbase * FEAT_DIM);
        uint4*       dst = (uint4*)As;
        #pragma unroll
        for (int i = tid; i < (QB * FEAT_DIM) / 8; i += 256) dst[i] = src[i];
    }
    if (tid < QB) { thr[tid] = -3.0e38f; scnt[tid] = 0; }
    __syncthreads();

    // ---- Per-row top-10 state (threads 0..15, registers) -------------------
    float tv[MAXK];
    int   ti[MAXK];
    #pragma unroll
    for (int j = 0; j < MAXK; ++j) { tv[j] = -3.0e38f; ti[j] = 0; }

    // A fragment base in LDS: lane holds row M=l15, 16 contiguous K at offset
    // lhalf*16 (matches B fragment K order).
    const __bf16* arow = As + l15 * FEAT_DIM + (lhalf << 4);

    // filter-phase coordinates: each thread owns 16 columns of one row
    const int frow = tid >> 4;
    const int fcol = tid & 15;

    // ---- Stream the bank ---------------------------------------------------
    for (int nbase = 0; nbase < BANK_SIZE; nbase += NB) {
        const int ncol = nbase + wave * 32;       // this wave's 32 bank rows

        // B fragments: lane holds column N, 16 contiguous K at offset
        // lhalf*16 (same K order as the A fragment).  Two tiles share one A.
        const __bf16* brow0 =
            mn + (size_t)(ncol + l15) * FEAT_DIM + (lhalf << 4);
        const __bf16* brow1 = brow0 + (size_t)16 * FEAT_DIM;

        // Prefetch next chunk's rows for this lane.
        if (nbase + NB < BANK_SIZE) {
            __builtin_prefetch(brow0 + (size_t)NB * FEAT_DIM, 0, 0);
            __builtin_prefetch(brow1 + (size_t)NB * FEAT_DIM, 0, 0);
        }

        v8f acc0 = {0.f, 0.f, 0.f, 0.f, 0.f, 0.f, 0.f, 0.f};
        v8f acc1 = {0.f, 0.f, 0.f, 0.f, 0.f, 0.f, 0.f, 0.f};

        #pragma unroll 2
        for (int k0 = 0; k0 < FEAT_DIM; k0 += 32) {
            union { v16bf v; uint4 q[2]; } au;
            au.q[0] = *(const uint4*)(arow + k0);        // K = k0 + lhalf*16 ..
            au.q[1] = *(const uint4*)(arow + k0 + 8);    // K = k0+8+lhalf*16 ..
            const v16bf b0 = *(const v16bf*)(brow0 + k0);
            const v16bf b1 = *(const v16bf*)(brow1 + k0);

            acc0 = __builtin_amdgcn_wmma_f32_16x16x32_bf16(
                false, au.v, false, b0, (short)0, acc0, false, false);
            // identical opcode, same A VGPRs -> reuse_a hint is legal
            acc1 = __builtin_amdgcn_wmma_f32_16x16x32_bf16(
                false, au.v, false, b1, (short)0, acc1, true, false);
        }

        // C layout: VGPR i -> M = i + 8*lane_half, N = lane&15 (ISA 7.12.2).
        #pragma unroll
        for (int i = 0; i < 8; ++i) {
            sim[i + (lhalf << 3)][wave * 32 + l15]      = acc0[i];
            sim[i + (lhalf << 3)][wave * 32 + 16 + l15] = acc1[i];
        }
        __syncthreads();

        // ---- Threshold filter: 256 threads, 16 cols each -------------------
        {
            const float t = thr[frow];
            #pragma unroll 4
            for (int c = fcol; c < NB; c += 16) {
                const float v = sim[frow][c];
                if (v > t) {
                    const int pos = atomicAdd(&scnt[frow], 1);   // ds_add_rtn
                    if (pos < CAP) {
                        candv[frow][pos] = v;
                        candi[frow][pos] = nbase + c;
                    }
                }
            }
        }
        __syncthreads();

        // ---- Sorted insertion on survivors (threads 0..15) -----------------
        if (tid < QB) {
            const int n = scnt[tid];
            if (n > CAP) {
                // overflow (first chunk / pathological): full scan fallback
                for (int c = 0; c < NB; ++c) {
                    const float v = sim[tid][c];
                    if (v > tv[MAXK - 1]) {
                        int j = MAXK - 1;
                        while (j > 0 && tv[j - 1] < v) {
                            tv[j] = tv[j - 1]; ti[j] = ti[j - 1]; --j;
                        }
                        tv[j] = v; ti[j] = nbase + c;
                    }
                }
            } else {
                for (int c = 0; c < n; ++c) {
                    const float v = candv[tid][c];
                    if (v > tv[MAXK - 1]) {
                        int j = MAXK - 1;
                        while (j > 0 && tv[j - 1] < v) {
                            tv[j] = tv[j - 1]; ti[j] = ti[j - 1]; --j;
                        }
                        tv[j] = v; ti[j] = candi[tid][c];
                    }
                }
            }
            thr[tid]  = tv[MAXK - 1];
            scnt[tid] = 0;
        }
        __syncthreads();
    }

    // ---- Adaptive k + masked softmax (threads 0..15) -----------------------
    if (tid < QB) {
        const float* p = preds + (size_t)(qbase + tid) * NUM_CLASSES;
        float conf = 0.f;
        #pragma unroll
        for (int c = 0; c < NUM_CLASSES; ++c) {
            const float s = 1.0f / (1.0f + __expf(-p[c]));
            conf += fabsf(s - 0.5f);
        }
        conf *= (1.0f / (float)NUM_CLASSES);
        int k = (int)(1.0f + 9.0f * (1.0f - conf));   // floor, k in [1,10]
        if (k > MAXK) k = MAXK;
        if (k < 1) k = 1;

        const float mx = tv[0];                       // list sorted descending
        float w[MAXK];
        float sum = 0.f;
        #pragma unroll
        for (int j = 0; j < MAXK; ++j) {
            const float e = (j < k) ? __expf(tv[j] - mx) : 0.0f;
            w[j] = e;
            sum += e;
        }
        const float inv = 1.0f / sum;
        #pragma unroll
        for (int j = 0; j < MAXK; ++j) {
            wbuf[tid][j] = w[j] * inv;
            ibuf[tid][j] = ti[j];
        }
    }
    __syncthreads();

    // ---- Weighted gather of raw f32 bank rows -> output --------------------
    // 1024 dims = 256 float4 per row; one float4 per thread per query row.
    const int d4 = tid;
    for (int r = 0; r < QB; ++r) {
        float wj[MAXK];
        int   ix[MAXK];
        #pragma unroll
        for (int j = 0; j < MAXK; ++j) { wj[j] = wbuf[r][j]; ix[j] = ibuf[r][j]; }

        float4 acc = {0.f, 0.f, 0.f, 0.f};
        #pragma unroll
        for (int j = 0; j < MAXK; ++j) {
            const float4 m4 =
                ((const float4*)(memory + (size_t)ix[j] * FEAT_DIM))[d4];
            acc.x += wj[j] * m4.x;
            acc.y += wj[j] * m4.y;
            acc.z += wj[j] * m4.z;
            acc.w += wj[j] * m4.w;
        }
        ((float4*)(out + (size_t)(qbase + r) * FEAT_DIM))[d4] = acc;
    }
}

// ---------------------------------------------------------------------------
// Launch: inputs are (query, predictions, memory) per setup_inputs() order.
// Workspace layout: [0, 8MB)    query bf16 (4096*1024*2)
//                   [8MB,136MB) bank  bf16 (65536*1024*2)
// ---------------------------------------------------------------------------
extern "C" void kernel_launch(void* const* d_in, const int* in_sizes, int n_in,
                              void* d_out, int out_size, void* d_ws, size_t ws_size,
                              hipStream_t stream) {
    const float* query  = (const float*)d_in[0];
    const float* preds  = (const float*)d_in[1];
    const float* memory = (const float*)d_in[2];
    float*       out    = (float*)d_out;

    __bf16* qn = (__bf16*)d_ws;
    __bf16* mn = (__bf16*)((char*)d_ws + (size_t)BATCH * FEAT_DIM * sizeof(__bf16));

    lamb_normalize_bf16<<<BATCH, 256, 0, stream>>>(query, qn);
    lamb_normalize_bf16<<<BANK_SIZE, 256, 0, stream>>>(memory, mn);
    lamb_sim_topk<<<BATCH / QB, 256, 0, stream>>>(qn, mn, memory, preds, out);
}
